// Attention_47467978556096
// MI455X (gfx1250) — matrix operand order repaired and checked by reading, not compiled, not run on hardware
//
#include <hip/hip_runtime.h>

typedef _Float16 h8   __attribute__((ext_vector_type(8)));
typedef _Float16 h16  __attribute__((ext_vector_type(16)));
typedef float    f8   __attribute__((ext_vector_type(8)));
typedef float    f4v  __attribute__((ext_vector_type(4)));
typedef unsigned int u32x4 __attribute__((ext_vector_type(4)));
typedef int      i32x4 __attribute__((ext_vector_type(4)));
typedef int      i32x8 __attribute__((ext_vector_type(8)));

#define HEADS  16
#define DHEAD  64
#define DMODEL 1024
#define BATCH  2
#define SEQ    2048

#define LOG2E  1.44269504088896340f

// ---------------------------------------------------------------------------
// Fragment loaders following CDNA5 WMMA VGPR layouts (CDNA5 ISA):
//  A (16x32 f16): lane L: row M = L%16, h = L/16;
//     elements 0..7  -> K =  8*h + e ;  elements 8..15 -> K = 16 + 8*h + (e-8)
//  B (32x16 f16): lane L: col N = L%16, h = L/16; same K order as A
//  C/D (16x16 f32): vgpr i, lane L: row M = i + 8*(L/16), col N = L%16
// ---------------------------------------------------------------------------
__device__ __forceinline__ h16 load_a_frag(const _Float16* row_ptr, int h) {
  h8 lo = *(const h8*)(row_ptr + 8 * h);
  h8 hi = *(const h8*)(row_ptr + 16 + 8 * h);
  h16 r;
#pragma unroll
  for (int i = 0; i < 8; ++i) { r[i] = lo[i]; r[8 + i] = hi[i]; }
  return r;
}

__device__ __forceinline__ h16 load_b_frag(const _Float16* colT_ptr, int h) {
  h8 lo = *(const h8*)(colT_ptr + 8 * h);
  h8 hi = *(const h8*)(colT_ptr + 16 + 8 * h);
  h16 r;
#pragma unroll
  for (int i = 0; i < 8; ++i) { r[i] = lo[i]; r[8 + i] = hi[i]; }
  return r;
}

// ---------------------------------------------------------------------------
// Tensor Data Mover: 2-D tiled load global -> LDS, 64 rows x 64 f16 per row,
// with LDS padding +4 DWORDs per 32-DWORD row => 72-half row stride
// (bank-conflict-free fragment reads). D# packing per CDNA5 ISA.
// ---------------------------------------------------------------------------
__device__ __forceinline__ void tdm_load_tile64x64(unsigned lds_off,
                                                   const void* gptr,
                                                   unsigned tensor_dim0,
                                                   unsigned tensor_dim1,
                                                   unsigned long long stride0) {
  unsigned long long ga = (unsigned long long)(uintptr_t)gptr;
  u32x4 g0;
  g0[0] = 1u;                                    // count=1 (valid user D#)
  g0[1] = lds_off;                               // lds_addr (bytes)
  g0[2] = (unsigned)ga;                          // global_addr[31:0]
  g0[3] = ((unsigned)(ga >> 32) & 0x01FFFFFFu)   // global_addr[56:32]
          | (2u << 30);                          // type=2 ("image")
  i32x8 g1;
  g1[0] = (int)((1u << 16)                       // data_size = 2 bytes
              | (1u << 20)                       // pad_enable
              | (4u << 22)                       // pad_interval: 32 DWORDs
              | (3u << 25));                     // pad_amount:   4 DWORDs
  g1[1] = (int)((tensor_dim0 & 0xFFFFu) << 16);  // tensor_dim0[15:0] @ bit48
  g1[2] = (int)((tensor_dim0 >> 16) | ((tensor_dim1 & 0xFFFFu) << 16));
  g1[3] = (int)((tensor_dim1 >> 16) | (64u << 16));  // tile_dim0 = 64
  g1[4] = 64;                                        // tile_dim1 = 64
  g1[5] = (int)(unsigned)stride0;                    // dim0_stride lo
  g1[6] = (int)((unsigned)(stride0 >> 32) & 0xFFFFu);// hi; dim1_stride = 0
  g1[7] = 0;
  i32x4 z4 = {0, 0, 0, 0};
#if defined(__clang_major__) && (__clang_major__ >= 23)
  i32x8 z8 = {0, 0, 0, 0, 0, 0, 0, 0};
  __builtin_amdgcn_tensor_load_to_lds(g0, g1, z4, z4, z8, 0);
#else
  __builtin_amdgcn_tensor_load_to_lds(g0, g1, z4, z4, 0);
#endif
}

// ---------------------------------------------------------------------------
// Projection GEMM: Out (f16) = X[4096,1024](f32) @ W[1024,1024](f32)
// Block tile 128x128, BK=32, LDS double-buffered (1 barrier / K-step).
// TR==0: Out[((b*H+h)*S+s)*64+d]    TR==1: Out[((b*H+h)*64+d)*S+s] (for V)
// ---------------------------------------------------------------------------
template <int TR>
__global__ __launch_bounds__(256)
void proj_gemm_f16(const float* __restrict__ X, const float* __restrict__ W,
                   _Float16* __restrict__ Out) {
  __shared__ _Float16 As[2][128][40];
  __shared__ _Float16 BsT[2][128][40];

  const int tid  = threadIdx.x;
  const int wave = tid >> 5;
  const int lane = tid & 31;
  const int half = lane >> 4;
  const int l16  = lane & 15;
  const int wm   = wave & 1;
  const int wn   = wave >> 1;
  const int blockM = blockIdx.y * 128;
  const int blockN = blockIdx.x * 128;

  f8 acc[4][2];
#pragma unroll
  for (int mt = 0; mt < 4; ++mt)
#pragma unroll
    for (int nt = 0; nt < 2; ++nt)
#pragma unroll
      for (int i = 0; i < 8; ++i) acc[mt][nt][i] = 0.0f;

  f4v aReg[4], bReg[4];

  auto load_global = [&](int k0) {
#pragma unroll
    for (int i = 0; i < 4; ++i) {
      int lin  = tid + 256 * i;
      int arow = lin >> 3, ac = (lin & 7) * 4;
      aReg[i] = *(const f4v*)(X + (size_t)(blockM + arow) * DMODEL + k0 + ac);
      int brow = lin >> 5, bc = (lin & 31) * 4;
      bReg[i] = *(const f4v*)(W + (size_t)(k0 + brow) * DMODEL + blockN + bc);
    }
  };
  auto store_lds = [&](int buf) {
#pragma unroll
    for (int i = 0; i < 4; ++i) {
      int lin  = tid + 256 * i;
      int arow = lin >> 3, ac = (lin & 7) * 4;
      int brow = lin >> 5, bc = (lin & 31) * 4;
#pragma unroll
      for (int j = 0; j < 4; ++j) {
        As[buf][arow][ac + j]  = (_Float16)aReg[i][j];
        BsT[buf][bc + j][brow] = (_Float16)bReg[i][j];
      }
    }
  };

  load_global(0);
  store_lds(0);

  constexpr int KSTEPS = DMODEL / 32;
  for (int step = 0; step < KSTEPS; ++step) {
    __syncthreads();
    const int buf = step & 1;
    if (step + 1 < KSTEPS) load_global((step + 1) * 32);

    h16 afrag[4], bfrag[2];
#pragma unroll
    for (int mt = 0; mt < 4; ++mt)
      afrag[mt] = load_a_frag(&As[buf][wm * 64 + mt * 16 + l16][0], half);
#pragma unroll
    for (int nt = 0; nt < 2; ++nt)
      bfrag[nt] = load_b_frag(&BsT[buf][wn * 32 + nt * 16 + l16][0], half);
#pragma unroll
    for (int mt = 0; mt < 4; ++mt)
#pragma unroll
      for (int nt = 0; nt < 2; ++nt)
        acc[mt][nt] = __builtin_amdgcn_wmma_f32_16x16x32_f16(
            false, afrag[mt], false, bfrag[nt], (short)0, acc[mt][nt],
            false, false);

    if (step + 1 < KSTEPS) store_lds(buf ^ 1);
  }

#pragma unroll
  for (int mt = 0; mt < 4; ++mt)
#pragma unroll
    for (int nt = 0; nt < 2; ++nt)
#pragma unroll
      for (int i = 0; i < 8; ++i) {
        int m = blockM + wm * 64 + mt * 16 + i + 8 * half;
        int n = blockN + wn * 32 + nt * 16 + l16;
        int b = m >> 11, s = m & (SEQ - 1);
        int h = n >> 6,  d = n & (DHEAD - 1);
        size_t idx;
        if (TR)
          idx = (((size_t)b * HEADS + h) * DHEAD + d) * SEQ + s;
        else
          idx = (((size_t)b * HEADS + h) * SEQ + s) * DHEAD + d;
        Out[idx] = (_Float16)acc[mt][nt][i];
      }
}

// ---------------------------------------------------------------------------
// Flash attention: per (b,h), 64-query tiles; K/V blocks streamed by the TDM
// into double-buffered LDS, one workgroup barrier per key block.
// Softmax runs in the exp2 domain (single v_exp_f32 per value, co-executes
// with WMMA as a TRANS op). 128 threads = 4 waves, 16 query rows/wave.
// ---------------------------------------------------------------------------
__global__ __launch_bounds__(128)
void flash_attn(const _Float16* __restrict__ Qh, const _Float16* __restrict__ Kh,
                const _Float16* __restrict__ VhT, const float* __restrict__ mask,
                float* __restrict__ Out) {
  __shared__ _Float16 Ks[2][64][72];     // [key][dim]  (TDM-padded stride 72)
  __shared__ _Float16 VsT[2][64][72];    // [dim][key]  (V pre-transposed)
  __shared__ _Float16 Ps[4][16][72];     // per-wave P tile [qrow][key]

  const int tid  = threadIdx.x;
  const int wave = tid >> 5;
  const int lane = tid & 31;
  const int half = lane >> 4;
  const int l16  = lane & 15;
  const int bh = blockIdx.y;
  const int b  = bh >> 4;
  const int hh = bh & 15;
  const int q0 = blockIdx.x * 64;
  const size_t base = (size_t)bh * SEQ * DHEAD;

  const _Float16* Kbh = Kh + base;       // [key][64]
  const _Float16* Vbh = VhT + base;      // [dim][2048]

  auto issue_tiles = [&](int kb, int buf) {
    tdm_load_tile64x64((unsigned)(uintptr_t)&Ks[buf][0][0],
                       Kbh + (size_t)kb * 64 * DHEAD, DHEAD, 64, DHEAD);
    tdm_load_tile64x64((unsigned)(uintptr_t)&VsT[buf][0][0],
                       Vbh + (size_t)kb * 64, SEQ, DHEAD, SEQ);
  };

  h16 qfrag[2];
  {
    const _Float16* qp = Qh + base + (size_t)(q0 + wave * 16 + l16) * DHEAD;
    qfrag[0] = load_a_frag(qp, half);
    qfrag[1] = load_a_frag(qp + 32, half);
  }

  f8 o[4];
#pragma unroll
  for (int t = 0; t < 4; ++t)
#pragma unroll
    for (int i = 0; i < 8; ++i) o[t][i] = 0.0f;
  float m_i[8], l_i[8];                  // running max in log2 domain
#pragma unroll
  for (int i = 0; i < 8; ++i) { m_i[i] = -1e30f; l_i[i] = 0.0f; }

  const float scaleL2 = 0.125f * LOG2E;  // (1/sqrt(64)) * log2(e)
  constexpr int NBLK = SEQ / 64;

  if (wave == 0) issue_tiles(0, 0);

  for (int kb = 0; kb < NBLK; ++kb) {
    const int cur = kb & 1;
    if (wave == 0) __builtin_amdgcn_s_wait_tensorcnt(0);
    __syncthreads();                     // barrier A: buffers[cur] visible,
                                         // and gates TDM reuse of buf cur^1
    if (wave == 0 && kb + 1 < NBLK) issue_tiles(kb + 1, cur ^ 1);

    // S = Q @ K^T : batch fragment loads per pair of N-tiles so the
    // scheduler can stagger s_wait_dscnt under WMMA issue.
    f8 sacc[4];
#pragma unroll
    for (int ng = 0; ng < 2; ++ng) {
      h16 kf[2][2];
#pragma unroll
      for (int j = 0; j < 2; ++j)
#pragma unroll
        for (int ks = 0; ks < 2; ++ks)
          kf[j][ks] =
              load_b_frag(&Ks[cur][(ng * 2 + j) * 16 + l16][ks * 32], half);
#pragma unroll
      for (int j = 0; j < 2; ++j) {
        const int nt = ng * 2 + j;
#pragma unroll
        for (int i = 0; i < 8; ++i) sacc[nt][i] = 0.0f;
        sacc[nt] = __builtin_amdgcn_wmma_f32_16x16x32_f16(
            false, qfrag[0], false, kf[j][0], (short)0, sacc[nt], false, false);
        sacc[nt] = __builtin_amdgcn_wmma_f32_16x16x32_f16(
            false, qfrag[1], false, kf[j][1], (short)0, sacc[nt], false, false);
      }
    }

    const int kbase = kb * 64;
    float biasL2[4];
#pragma unroll
    for (int nt = 0; nt < 4; ++nt)
      biasL2[nt] = -(1.0f - mask[b * SEQ + kbase + nt * 16 + l16]) *
                   (1e12f * LOG2E);

    float pv[4][8];                      // log2-domain scores
#pragma unroll
    for (int nt = 0; nt < 4; ++nt)
#pragma unroll
      for (int i = 0; i < 8; ++i)
        pv[nt][i] = fmaf(sacc[nt][i], scaleL2, biasL2[nt]);

    // Online softmax (exp2 domain). Row r = i + 8*half spans 16 lanes.
    float mnew[8], fsc[8];
#pragma unroll
    for (int i = 0; i < 8; ++i) {
      float mx = fmaxf(fmaxf(pv[0][i], pv[1][i]), fmaxf(pv[2][i], pv[3][i]));
#pragma unroll
      for (int d = 1; d < 16; d <<= 1) mx = fmaxf(mx, __shfl_xor(mx, d, 32));
      mnew[i] = fmaxf(m_i[i], mx);
      fsc[i]  = __builtin_amdgcn_exp2f(m_i[i] - mnew[i]);
    }
#pragma unroll
    for (int nt = 0; nt < 4; ++nt)
#pragma unroll
      for (int i = 0; i < 8; ++i)
        pv[nt][i] = __builtin_amdgcn_exp2f(pv[nt][i] - mnew[i]);
#pragma unroll
    for (int i = 0; i < 8; ++i) {
      float s = pv[0][i] + pv[1][i] + pv[2][i] + pv[3][i];
#pragma unroll
      for (int d = 1; d < 16; d <<= 1) s += __shfl_xor(s, d, 32);
      l_i[i] = l_i[i] * fsc[i] + s;
      m_i[i] = mnew[i];
    }
#pragma unroll
    for (int t = 0; t < 4; ++t)
#pragma unroll
      for (int i = 0; i < 8; ++i) o[t][i] *= fsc[i];

    // P (C layout) -> LDS -> A-fragment layout. Ps[wave] is wave-private:
    // only the wave's own DS ops must complete (no workgroup barrier).
#pragma unroll
    for (int nt = 0; nt < 4; ++nt)
#pragma unroll
      for (int i = 0; i < 8; ++i)
        Ps[wave][i + 8 * half][nt * 16 + l16] = (_Float16)pv[nt][i];
    asm volatile("s_wait_dscnt 0x0" ::: "memory");

    h16 pfrag[2];
    pfrag[0] = load_a_frag(&Ps[wave][l16][0], half);
    pfrag[1] = load_a_frag(&Ps[wave][l16][32], half);
#pragma unroll
    for (int ng = 0; ng < 2; ++ng) {
      h16 vf[2][2];
#pragma unroll
      for (int j = 0; j < 2; ++j)
#pragma unroll
        for (int ks = 0; ks < 2; ++ks)
          vf[j][ks] =
              load_b_frag(&VsT[cur][(ng * 2 + j) * 16 + l16][ks * 32], half);
#pragma unroll
      for (int j = 0; j < 2; ++j) {
        const int nt = ng * 2 + j;
        o[nt] = __builtin_amdgcn_wmma_f32_16x16x32_f16(
            false, pfrag[0], false, vf[j][0], (short)0, o[nt], false, false);
        o[nt] = __builtin_amdgcn_wmma_f32_16x16x32_f16(
            false, pfrag[1], false, vf[j][1], (short)0, o[nt], false, false);
      }
    }
  }

  // Normalize and store f32: Out[b, s, h*64 + d]
#pragma unroll
  for (int t = 0; t < 4; ++t)
#pragma unroll
    for (int i = 0; i < 8; ++i) {
      int s = q0 + wave * 16 + i + 8 * half;
      int d = t * 16 + l16;
      Out[((size_t)b * SEQ + s) * DMODEL + hh * DHEAD + d] = o[t][i] / l_i[i];
    }
}

// ---------------------------------------------------------------------------
extern "C" void kernel_launch(void* const* d_in, const int* in_sizes, int n_in,
                              void* d_out, int out_size, void* d_ws, size_t ws_size,
                              hipStream_t stream) {
  const float* q    = (const float*)d_in[0];
  const float* k    = (const float*)d_in[1];
  const float* v    = (const float*)d_in[2];
  const float* mask = (const float*)d_in[3];
  const float* Wq   = (const float*)d_in[4];
  const float* Wk   = (const float*)d_in[5];
  const float* Wv   = (const float*)d_in[6];

  const size_t per = (size_t)BATCH * HEADS * SEQ * DHEAD;  // 8MB per tensor
  _Float16* Qh  = (_Float16*)d_ws;
  _Float16* Kh  = Qh + per;
  _Float16* VhT = Kh + per;

  dim3 gProj(DMODEL / 128, (BATCH * SEQ) / 128);
  proj_gemm_f16<0><<<gProj, 256, 0, stream>>>(q, Wq, Qh);
  proj_gemm_f16<0><<<gProj, 256, 0, stream>>>(k, Wk, Kh);
  proj_gemm_f16<1><<<gProj, 256, 0, stream>>>(v, Wv, VhT);

  dim3 gAttn(SEQ / 64, BATCH * HEADS);
  flash_attn<<<gAttn, 128, 0, stream>>>(Qh, Kh, VhT, mask, (float*)d_out);
}
